// GNNMotion_76768245449447
// MI455X (gfx1250) — matrix operand order repaired and checked by reading, not compiled, not run on hardware
//
#include <hip/hip_runtime.h>
#include <hip/hip_bf16.h>

// ---------------- problem constants (from reference) ----------------
#define NN     4096
#define HH     128
#define HEADS  4
#define DH     128
#define QKV    (HEADS*DH)   // 512
#define EE     131072
#define BBATCH 64
#define TSTEPS 30

typedef __attribute__((ext_vector_type(16))) __bf16 v16bf;
typedef __attribute__((ext_vector_type(8)))  float  v8f;
typedef __attribute__((ext_vector_type(4)))  unsigned v4u;
typedef __attribute__((ext_vector_type(8)))  int      v8i;
typedef __attribute__((ext_vector_type(4)))  int      v4i;

union BFU { unsigned short u; __bf16 b; };

__device__ __forceinline__ unsigned short f2bfu(float f) {
  unsigned u = __float_as_uint(f);
  u = (u + 0x7FFFu + ((u >> 16) & 1u)) >> 16;   // RNE truncate to bf16
  return (unsigned short)u;
}
__device__ __forceinline__ __bf16 u2bf(unsigned short u) { BFU c; c.u = u; return c.b; }

__device__ __forceinline__ v8f wmma_bf16(v16bf a, v16bf b, v8f c) {
  // D = A(16x32 bf16) * B(32x16 bf16) + C(16x16 f32)
  return __builtin_amdgcn_wmma_f32_16x16x32_bf16(false, a, false, b, (short)0, c,
                                                 false, false);
}

// ---- A fragment: 16x32 bf16, row-major source, row stride ld (elements) ----
__device__ __forceinline__ v16bf load_a_frag(const unsigned short* p, int ld) {
  int lane = threadIdx.x & 31;
  const unsigned short* row = p + (lane & 15) * ld;
  int kb = (lane < 16) ? 0 : 16;
  v16bf a;
#pragma unroll
  for (int i = 0; i < 8; ++i) {
    unsigned w0 = *(const unsigned*)(row + kb + 2 * i);
    a[2 * i]     = u2bf((unsigned short)w0);
    a[2 * i + 1] = u2bf((unsigned short)(w0 >> 16));
  }
  return a;
}

// ---- B fragment from B^T row-major (N x K): per-lane contiguous K ----
__device__ __forceinline__ v16bf load_bT_frag(const unsigned short* p, int ld) {
  int lane = threadIdx.x & 31;
  const unsigned short* col = p + (lane & 15) * ld + ((lane < 16) ? 0 : 16);
  v16bf b;
#pragma unroll
  for (int i = 0; i < 8; ++i) {
    unsigned w = *(const unsigned*)(col + 2 * i);
    b[2 * i]     = u2bf((unsigned short)w);
    b[2 * i + 1] = u2bf((unsigned short)(w >> 16));
  }
  return b;
}

// =====================================================================
// TDM: async 2-D tile load (global -> LDS). D# per CDNA5 ISA.
//   tile_w elements contiguous (data_size=2B), tile_h rows, row stride
//   stride_elems. All inputs wave-uniform -> SGPR descriptor groups.
// =====================================================================
__device__ __forceinline__ void tdm_load_2d(unsigned lds_off, const void* gptr,
                                            unsigned tile_w, unsigned tile_h,
                                            unsigned stride_elems) {
  unsigned long long ga = (unsigned long long)(size_t)gptr;
  const unsigned td1 = 1u << 20;            // generous dim-1 bound (no OOB clip)
  v4u g0;
  g0[0] = 1u;                                        // count=1, user descriptor
  g0[1] = lds_off;                                   // LDS byte address
  g0[2] = (unsigned)(ga & 0xFFFFFFFFull);            // global_addr[31:0]
  g0[3] = (unsigned)((ga >> 32) & 0x1FFFFFFull)      // global_addr[56:32]
        | (2u << 30);                                // type = 2 ("image")
  v8i g1;
  g1[0] = (int)(1u << 16);                           // data_size = 1 (2 bytes)
  g1[1] = (int)((stride_elems & 0xFFFFu) << 16);     // tensor_dim0[15:0]
  g1[2] = (int)(((stride_elems >> 16) & 0xFFFFu)     // tensor_dim0[31:16]
        | ((td1 & 0xFFFFu) << 16));                  // tensor_dim1[15:0]
  g1[3] = (int)(((td1 >> 16) & 0xFFFFu)              // tensor_dim1[31:16]
        | ((tile_w & 0xFFFFu) << 16));               // tile_dim0
  g1[4] = (int)(tile_h & 0xFFFFu);                   // tile_dim1 (tile_dim2=0)
  g1[5] = (int)stride_elems;                         // tensor_dim0_stride[31:0]
  g1[6] = 0;                                         // stride[47:32] | dim1_stride lo
  g1[7] = 0;
  v4i g2 = {0, 0, 0, 0};
  v4i g3 = {0, 0, 0, 0};
#if defined(__clang_major__) && (__clang_major__ >= 23)
  v8i g4 = {0, 0, 0, 0, 0, 0, 0, 0};
  __builtin_amdgcn_tensor_load_to_lds(g0, g1, g2, g3, g4, 0);
#else
  __builtin_amdgcn_tensor_load_to_lds(g0, g1, g2, g3, 0);
#endif
}

// =====================================================================
// Generic bf16 WMMA GEMM: out = act(A[MxK] * Bt^T + bias + resid)
//   grid = (M/16, N/64), block = 32 (one wave); K % 32 == 0.
// =====================================================================
__global__ void gemm_bf16_kernel(const unsigned short* __restrict__ A,
                                 const unsigned short* __restrict__ Bt,
                                 const float* __restrict__ bias,
                                 const float* __restrict__ resid,
                                 float* __restrict__ outF,
                                 unsigned short* __restrict__ outBf,
                                 int M, int N, int K, int do_relu) {
  int row0 = blockIdx.x * 16;
  int col0 = blockIdx.y * 64;
  v8f acc[4];
#pragma unroll
  for (int t = 0; t < 4; ++t)
#pragma unroll
    for (int i = 0; i < 8; ++i) acc[t][i] = 0.0f;

  for (int k0 = 0; k0 < K; k0 += 32) {
    if (k0 + 32 < K) {  // hint next A/B fragments -> global_prefetch_b8
      __builtin_prefetch(A + (size_t)row0 * K + k0 + 32, 0, 3);
      __builtin_prefetch(Bt + (size_t)col0 * K + k0 + 32, 0, 3);
    }
    v16bf a = load_a_frag(A + (size_t)row0 * K + k0, K);
#pragma unroll
    for (int t = 0; t < 4; ++t) {
      v16bf b = load_bT_frag(Bt + (size_t)(col0 + t * 16) * K + k0, K);
      acc[t] = wmma_bf16(a, b, acc[t]);
    }
  }

  int lane = threadIdx.x & 31;
  int n = lane & 15;
  int mb = (lane < 16) ? 0 : 8;
#pragma unroll
  for (int t = 0; t < 4; ++t) {
    int col = col0 + t * 16 + n;
    float bv = bias ? bias[col] : 0.0f;
#pragma unroll
    for (int i = 0; i < 8; ++i) {
      int row = row0 + mb + i;
      float v = acc[t][i] + bv;
      if (resid) v += resid[(size_t)row * N + col];
      if (do_relu) v = fmaxf(v, 0.0f);
      if (outF)  outF[(size_t)row * N + col] = v;
      if (outBf) outBf[(size_t)row * N + col] = f2bfu(v);
    }
  }
}

// =====================================================================
// Fused masked GAT attention (flash style, online softmax) with
// TDM double-buffered K/V tiles in LDS.
//   grid = (NN/16, HEADS), block = 32 (one wave per (head, 16-q-row tile)).
// =====================================================================
__global__ void gat_attn_kernel(const unsigned short* __restrict__ Qbf,
                                const unsigned short* __restrict__ Kbf,
                                const unsigned short* __restrict__ Vt,
                                const unsigned* __restrict__ maskbits,
                                unsigned short* __restrict__ attn_bf) {
  __shared__ unsigned short Klds[2][32 * DH];   // [node-in-tile][d]   8KB x2
  __shared__ unsigned short Vlds[2][DH * 32];   // [d][node-in-tile]   8KB x2
  __shared__ unsigned short Plds[16 * 32];      // P tile C->A layout bounce

  const int qt = blockIdx.x;
  const int h  = blockIdx.y;
  const int row0 = qt * 16;
  const int lane = threadIdx.x & 31;
  const int n  = lane & 15;
  const int mb = (lane < 16) ? 0 : 8;
  const float scale = 0.08838834764831845f;   // 1/sqrt(128)
  const float NEG = -__builtin_inff();

  // Q fragments for all 4 K-chunks of DH=128 (loaded once, reused all tiles)
  v16bf qf[4];
#pragma unroll
  for (int c = 0; c < 4; ++c)
    qf[c] = load_a_frag(Qbf + (size_t)row0 * QKV + h * DH + c * 32, QKV);

  v8f o[8];
#pragma unroll
  for (int t = 0; t < 8; ++t)
#pragma unroll
    for (int i = 0; i < 8; ++i) o[t][i] = 0.0f;
  float mrow[8], lrow[8];
#pragma unroll
  for (int i = 0; i < 8; ++i) { mrow[i] = NEG; lrow[i] = 0.0f; }

  // ---- TDM prologue: bring tile 0 into buffer 0 ----
  tdm_load_2d((unsigned)(size_t)&Klds[0][0],
              Kbf + (size_t)0 * QKV + h * DH, DH, 32, QKV);
  tdm_load_2d((unsigned)(size_t)&Vlds[0][0],
              Vt + (size_t)h * DH * NN + 0, 32, DH, NN);

  for (int it = 0; it < NN / 32; ++it) {
    const int cb = it * 32;
    const int cur = it & 1;
    if (cb + 32 < NN) {   // issue next tile into other buffer, then wait cur
      tdm_load_2d((unsigned)(size_t)&Klds[cur ^ 1][0],
                  Kbf + (size_t)(cb + 32) * QKV + h * DH, DH, 32, QKV);
      tdm_load_2d((unsigned)(size_t)&Vlds[cur ^ 1][0],
                  Vt + (size_t)h * DH * NN + (cb + 32), 32, DH, NN);
      __builtin_amdgcn_s_wait_tensorcnt(2);
    } else {
      __builtin_amdgcn_s_wait_tensorcnt(0);
    }

    // ---- scores for 32 key-columns from LDS K tile: two 16x16 tiles ----
    v8f s0, s1;
#pragma unroll
    for (int i = 0; i < 8; ++i) { s0[i] = 0.0f; s1[i] = 0.0f; }
#pragma unroll
    for (int c = 0; c < 4; ++c) {
      v16bf kf0 = load_bT_frag(&Klds[cur][c * 32], DH);
      v16bf kf1 = load_bT_frag(&Klds[cur][16 * DH + c * 32], DH);
      s0 = wmma_bf16(qf[c], kf0, s0);
      s1 = wmma_bf16(qf[c], kf1, s1);
    }

    // ---- scale + mask (one mask word covers both 16-col tiles) ----
    float sv0[8], sv1[8];
#pragma unroll
    for (int i = 0; i < 8; ++i) {
      int row = row0 + mb + i;
      unsigned w = maskbits[(size_t)row * (NN / 32) + (cb >> 5)];
      sv0[i] = ((w >> n) & 1u)        ? s0[i] * scale : NEG;
      sv1[i] = ((w >> (16 + n)) & 1u) ? s1[i] * scale : NEG;
    }

    // ---- online softmax: row reduce across the 16 lanes of each half ----
    float tmax[8];
#pragma unroll
    for (int i = 0; i < 8; ++i) tmax[i] = fmaxf(sv0[i], sv1[i]);
    for (int d = 1; d < 16; d <<= 1)
#pragma unroll
      for (int i = 0; i < 8; ++i)
        tmax[i] = fmaxf(tmax[i], __shfl_xor(tmax[i], d, 32));

    float cf[8], p0[8], p1[8], psum[8];
#pragma unroll
    for (int i = 0; i < 8; ++i) {
      float mnew = fmaxf(mrow[i], tmax[i]);
      cf[i] = (mrow[i] == NEG) ? 0.0f : __expf(mrow[i] - mnew);
      p0[i] = (sv0[i] == NEG) ? 0.0f : __expf(sv0[i] - mnew);
      p1[i] = (sv1[i] == NEG) ? 0.0f : __expf(sv1[i] - mnew);
      psum[i] = p0[i] + p1[i];
      mrow[i] = mnew;
    }
    for (int d = 1; d < 16; d <<= 1)
#pragma unroll
      for (int i = 0; i < 8; ++i) psum[i] += __shfl_xor(psum[i], d, 32);
#pragma unroll
    for (int i = 0; i < 8; ++i) lrow[i] = lrow[i] * cf[i] + psum[i];
#pragma unroll
    for (int t = 0; t < 8; ++t)
#pragma unroll
      for (int i = 0; i < 8; ++i) o[t][i] *= cf[i];

    // ---- P (16x32) through LDS: C-layout -> A-layout ----
#pragma unroll
    for (int i = 0; i < 8; ++i) {
      Plds[(mb + i) * 32 + n]      = f2bfu(p0[i]);
      Plds[(mb + i) * 32 + 16 + n] = f2bfu(p1[i]);
    }
    __syncthreads();                 // single-wave block: orders DS ops
    v16bf pf = load_a_frag(Plds, 32);
    __syncthreads();

    // ---- O += P @ V from LDS V tile ([d][node] -> contiguous K) ----
#pragma unroll
    for (int t = 0; t < 8; ++t) {
      v16bf vf = load_bT_frag(&Vlds[cur][t * 16 * 32], 32);
      o[t] = wmma_bf16(pf, vf, o[t]);
    }
  }

  // ---- finalize: divide by l, store bf16 [node][h*DH + d] ----
#pragma unroll
  for (int t = 0; t < 8; ++t)
#pragma unroll
    for (int i = 0; i < 8; ++i) {
      int row = row0 + mb + i;
      float denom = lrow[i];
      float v = (denom > 0.0f) ? o[t][i] / denom : 0.0f;
      attn_bf[(size_t)row * QKV + h * DH + t * 16 + n] = f2bfu(v);
    }
}

// ================= small scalar helper kernels =================
__global__ void enc1_kernel(const float* __restrict__ X, const float* __restrict__ w1,
                            const float* __restrict__ b1, unsigned short* __restrict__ outBf) {
  int idx = blockIdx.x * blockDim.x + threadIdx.x;       // NN*HH
  if (idx >= NN * HH) return;
  int nd = idx / HH, c = idx % HH;
  float acc = b1[c];
#pragma unroll
  for (int j = 0; j < 8; ++j) acc += X[nd * 8 + j] * w1[j * HH + c];
  outBf[idx] = f2bfu(fmaxf(acc, 0.0f));
}

__global__ void convert_wT_kernel(const float* __restrict__ W, unsigned short* __restrict__ WtBf,
                                  int K, int Nc) {
  int idx = blockIdx.x * blockDim.x + threadIdx.x;       // K*Nc
  if (idx >= K * Nc) return;
  int k = idx / Nc, nc = idx % Nc;
  WtBf[(size_t)nc * K + k] = f2bfu(W[idx]);
}

__global__ void zero_u32_kernel(unsigned* __restrict__ p, int nwords) {
  int idx = blockIdx.x * blockDim.x + threadIdx.x;
  if (idx < nwords) p[idx] = 0u;
}

__global__ void build_mask_kernel(const int* __restrict__ EI, unsigned* __restrict__ maskbits) {
  int e = blockIdx.x * blockDim.x + threadIdx.x;
  if (e >= EE) return;
  int s = EI[e];          // src row
  int d = EI[EE + e];     // dst col
  atomicOr(&maskbits[(size_t)s * (NN / 32) + (d >> 5)], 1u << (d & 31));
}

__global__ void transpose_v_kernel(const unsigned short* __restrict__ Vbf,
                                   unsigned short* __restrict__ Vt) {
  int idx = blockIdx.x * blockDim.x + threadIdx.x;       // NN*QKV
  if (idx >= NN * QKV) return;
  int nd = idx / QKV, hd = idx % QKV;
  int h = hd / DH, d = hd % DH;
  Vt[(size_t)h * DH * NN + (size_t)d * NN + nd] = Vbf[idx];
}

__global__ void gather_tgt_kernel(const float* __restrict__ Hn, const int* __restrict__ tgt,
                                  unsigned short* __restrict__ outBf) {
  int idx = blockIdx.x * blockDim.x + threadIdx.x;       // BBATCH*HH
  if (idx >= BBATCH * HH) return;
  int b = idx / HH, c = idx % HH;
  outBf[idx] = f2bfu(Hn[(size_t)tgt[b] * HH + c]);
}

__global__ void dec2_kernel(const float* __restrict__ A /*64x256*/,
                            const float* __restrict__ W /*256x60*/,
                            const float* __restrict__ bias, float* __restrict__ out) {
  int idx = blockIdx.x * blockDim.x + threadIdx.x;       // 64*60
  if (idx >= BBATCH * TSTEPS * 2) return;
  int b = idx / (TSTEPS * 2), c = idx % (TSTEPS * 2);
  float acc = bias[c];
  for (int j = 0; j < 256; ++j) acc += A[b * 256 + j] * W[j * (TSTEPS * 2) + c];
  out[idx] = acc;
}

// =====================================================================
extern "C" void kernel_launch(void* const* d_in, const int* in_sizes, int n_in,
                              void* d_out, int out_size, void* d_ws, size_t ws_size,
                              hipStream_t stream) {
  const float* X        = (const float*)d_in[0];
  const int*   EI       = (const int*)d_in[2];
  const int*   tgt      = (const int*)d_in[3];
  const float* xenc_w1  = (const float*)d_in[4];
  const float* xenc_b1  = (const float*)d_in[5];
  const float* xenc_w2  = (const float*)d_in[6];
  const float* xenc_b2  = (const float*)d_in[7];
  const float* gq[2]    = { (const float*)d_in[12], (const float*)d_in[17] };
  const float* gk[2]    = { (const float*)d_in[13], (const float*)d_in[18] };
  const float* gv[2]    = { (const float*)d_in[14], (const float*)d_in[19] };
  const float* gpw[2]   = { (const float*)d_in[15], (const float*)d_in[20] };
  const float* gpb[2]   = { (const float*)d_in[16], (const float*)d_in[21] };
  const float* dec_w1   = (const float*)d_in[22];
  const float* dec_b1   = (const float*)d_in[23];
  const float* dec_w2   = (const float*)d_in[24];
  const float* dec_b2   = (const float*)d_in[25];
  float* out = (float*)d_out;

  // ---------------- workspace carve-up ----------------
  char* p = (char*)d_ws;
  auto alloc = [&](size_t bytes) -> char* {
    char* r = p; p += (bytes + 255) & ~(size_t)255; return r;
  };
  unsigned*       maskbits = (unsigned*)alloc((size_t)NN * NN / 8);
  unsigned short* tmp_bf   = (unsigned short*)alloc((size_t)NN * HH * 2);
  float*          HnF[2]   = { (float*)alloc((size_t)NN * HH * 4),
                               (float*)alloc((size_t)NN * HH * 4) };
  unsigned short* HnB[2]   = { (unsigned short*)alloc((size_t)NN * HH * 2),
                               (unsigned short*)alloc((size_t)NN * HH * 2) };
  unsigned short* Qbf      = (unsigned short*)alloc((size_t)NN * QKV * 2);
  unsigned short* Kbf      = (unsigned short*)alloc((size_t)NN * QKV * 2);
  unsigned short* Vbf      = (unsigned short*)alloc((size_t)NN * QKV * 2);
  unsigned short* Vt       = (unsigned short*)alloc((size_t)NN * QKV * 2);
  unsigned short* attn_bf  = (unsigned short*)alloc((size_t)NN * QKV * 2);
  unsigned short* w2T      = (unsigned short*)alloc((size_t)HH * HH * 2);
  unsigned short* qT[2]    = { (unsigned short*)alloc((size_t)HH * QKV * 2),
                               (unsigned short*)alloc((size_t)HH * QKV * 2) };
  unsigned short* kT[2]    = { (unsigned short*)alloc((size_t)HH * QKV * 2),
                               (unsigned short*)alloc((size_t)HH * QKV * 2) };
  unsigned short* vT[2]    = { (unsigned short*)alloc((size_t)HH * QKV * 2),
                               (unsigned short*)alloc((size_t)HH * QKV * 2) };
  unsigned short* pwT[2]   = { (unsigned short*)alloc((size_t)QKV * HH * 2),
                               (unsigned short*)alloc((size_t)QKV * HH * 2) };
  unsigned short* dw1T     = (unsigned short*)alloc((size_t)HH * 256 * 2);
  unsigned short* htgt_bf  = (unsigned short*)alloc((size_t)BBATCH * HH * 2);
  float*          dec1F    = (float*)alloc((size_t)BBATCH * 256 * 4);

  const int TPB = 256;
  auto blocks = [](int nb, int tpb) { return (nb + tpb - 1) / tpb; };

  // ---------------- weight conversion (f32 -> bf16 transposed) ----------------
  convert_wT_kernel<<<blocks(HH * HH, TPB), TPB, 0, stream>>>(xenc_w2, w2T, HH, HH);
  for (int l = 0; l < 2; ++l) {
    convert_wT_kernel<<<blocks(HH * QKV, TPB), TPB, 0, stream>>>(gq[l], qT[l], HH, QKV);
    convert_wT_kernel<<<blocks(HH * QKV, TPB), TPB, 0, stream>>>(gk[l], kT[l], HH, QKV);
    convert_wT_kernel<<<blocks(HH * QKV, TPB), TPB, 0, stream>>>(gv[l], vT[l], HH, QKV);
    convert_wT_kernel<<<blocks(QKV * HH, TPB), TPB, 0, stream>>>(gpw[l], pwT[l], QKV, HH);
  }
  convert_wT_kernel<<<blocks(HH * 256, TPB), TPB, 0, stream>>>(dec_w1, dw1T, HH, 256);

  // ---------------- adjacency bit-mask ----------------
  zero_u32_kernel<<<blocks(NN * NN / 32, TPB), TPB, 0, stream>>>(maskbits, NN * NN / 32);
  build_mask_kernel<<<blocks(EE, TPB), TPB, 0, stream>>>(EI, maskbits);

  // ---------------- node encoder ----------------
  enc1_kernel<<<blocks(NN * HH, TPB), TPB, 0, stream>>>(X, xenc_w1, xenc_b1, tmp_bf);
  gemm_bf16_kernel<<<dim3(NN / 16, HH / 64), 32, 0, stream>>>(
      tmp_bf, w2T, xenc_b2, nullptr, HnF[0], HnB[0], NN, HH, HH, 1);

  // ---------------- two GAT layers ----------------
  for (int l = 0; l < 2; ++l) {
    const unsigned short* Hin_bf = HnB[l & 1];
    const float*          Hin_f  = HnF[l & 1];
    float*          Hout_f  = HnF[(l + 1) & 1];
    unsigned short* Hout_bf = HnB[(l + 1) & 1];

    gemm_bf16_kernel<<<dim3(NN / 16, QKV / 64), 32, 0, stream>>>(
        Hin_bf, qT[l], nullptr, nullptr, nullptr, Qbf, NN, QKV, HH, 0);
    gemm_bf16_kernel<<<dim3(NN / 16, QKV / 64), 32, 0, stream>>>(
        Hin_bf, kT[l], nullptr, nullptr, nullptr, Kbf, NN, QKV, HH, 0);
    gemm_bf16_kernel<<<dim3(NN / 16, QKV / 64), 32, 0, stream>>>(
        Hin_bf, vT[l], nullptr, nullptr, nullptr, Vbf, NN, QKV, HH, 0);
    transpose_v_kernel<<<blocks(NN * QKV, TPB), TPB, 0, stream>>>(Vbf, Vt);

    gat_attn_kernel<<<dim3(NN / 16, HEADS), 32, 0, stream>>>(Qbf, Kbf, Vt, maskbits, attn_bf);

    gemm_bf16_kernel<<<dim3(NN / 16, HH / 64), 32, 0, stream>>>(
        attn_bf, pwT[l], gpb[l], Hin_f, Hout_f, Hout_bf, NN, HH, QKV, 1);
  }

  // ---------------- decoder ----------------
  gather_tgt_kernel<<<blocks(BBATCH * HH, TPB), TPB, 0, stream>>>(HnF[0], tgt, htgt_bf);
  gemm_bf16_kernel<<<dim3(BBATCH / 16, 256 / 64), 32, 0, stream>>>(
      htgt_bf, dw1T, dec_b1, nullptr, dec1F, nullptr, BBATCH, 256, HH, 1);
  dec2_kernel<<<blocks(BBATCH * TSTEPS * 2, TPB), TPB, 0, stream>>>(dec1F, dec_w2, dec_b2, out);

  (void)in_sizes; (void)n_in; (void)out_size; (void)ws_size;
}
